// TAHIN_52458730553630
// MI455X (gfx1250) — compile-verified, matrix operand order checked by reading, not run on hardware
//
#include <hip/hip_runtime.h>
#include <hip/hip_bf16.h>

// ---------------- problem constants ----------------
#define NU 100000
#define NI 100000
#define NT 200000          // NU + NI
#define DIM 64
#define NINT 128
#define HIDN 128
#define NE 1000000L
#define NMPE 500000L
#define BSZ 4096
#define NLAYERS 2

// ---------------- workspace layout (float offsets) ----------------
#define OFF_DEG    0L            // 200000  (reused as outdeg/indeg for graphconv)
#define OFF_DINV   200000L       // 200000
#define OFF_EMB    400000L       // 12,800,000
#define OFF_TOTAL  13200000L     // 12,800,000
#define OFF_GNN    26000000L     // 12,800,000 (reused: h2u @ OFF_GNN, h2i @ OFF_GNN+6.4M)
#define OFF_Z      38800000L     // zu 12.8M, zi 12.8M
#define OFF_E1N    64400000L     // 262144
#define OFF_E2N    64662144L     // 262144
#define OFF_E1B    64924288L     // 131072 floats = 262144 ushort
#define OFF_E2B    65055360L     // 131072 floats
#define OFF_RS     65186432L     // 4096
#define OFF_WATT   65190528L     // 4
#define OFF_BETA   65190536L     // 4

// ---------------- WMMA types / helpers ----------------
typedef __attribute__((ext_vector_type(16))) __bf16 v16bf;
typedef __attribute__((ext_vector_type(8)))  __bf16 v8bf;
typedef __attribute__((ext_vector_type(8)))  float  v8f;
typedef __attribute__((ext_vector_type(4)))  float  v4f_;

__device__ __forceinline__ unsigned short bf16bits(float f) {
    unsigned u = __builtin_bit_cast(unsigned, f);
    unsigned r = u + 0x7FFFu + ((u >> 16) & 1u);       // RNE
    return (unsigned short)(r >> 16);
}
__device__ __forceinline__ unsigned packbf2(float f0, float f1) {
    return (unsigned)bf16bits(f0) | ((unsigned)bf16bits(f1) << 16);
}
// Fragment = two contiguous 8-element bf16 runs -> two 16B loads + concat.
__device__ __forceinline__ v16bf frag16(const unsigned short* lo, const unsigned short* hi) {
    v8bf a = *(const v8bf*)lo;
    v8bf b = *(const v8bf*)hi;
    return __builtin_shufflevector(a, b, 0,1,2,3,4,5,6,7,8,9,10,11,12,13,14,15);
}

// ---- cross-lane butterflies via ds_swizzle_b32 (group-of-32, XOR mask) ----
// offset encoding: [14:10]=xor, [9:5]=or, [4:0]=and  -> xor_mask<<10 | 0x1F
template <int IMM>
__device__ __forceinline__ float swz(float v) {
    return __builtin_bit_cast(float,
        __builtin_amdgcn_ds_swizzle(__builtin_bit_cast(int, v), IMM));
}
__device__ __forceinline__ float sum16(float v) {     // reduce within 16-lane group
    v += swz<0x041F>(v);   // xor 1
    v += swz<0x081F>(v);   // xor 2
    v += swz<0x101F>(v);   // xor 4
    v += swz<0x201F>(v);   // xor 8
    return v;
}
__device__ __forceinline__ float max16(float v) {
    v = fmaxf(v, swz<0x041F>(v));
    v = fmaxf(v, swz<0x081F>(v));
    v = fmaxf(v, swz<0x101F>(v));
    v = fmaxf(v, swz<0x201F>(v));
    return v;
}
__device__ __forceinline__ float sum32(float v) {     // full-wave reduce
    v = sum16(v);
    v += swz<0x401F>(v);   // xor 16
    return v;
}

// ---------------- elementwise / scatter kernels ----------------
__global__ __launch_bounds__(256) void k_zero(float* p, long n) {
    long i = (long)blockIdx.x * blockDim.x + threadIdx.x;
    if (i < n) p[i] = 0.0f;
}

__global__ __launch_bounds__(256) void k_init_emb(const float* __restrict__ uf,
                                                  const float* __restrict__ itf,
                                                  float* emb, float* total) {
    long i = (long)blockIdx.x * blockDim.x + threadIdx.x;
    if (i >= (long)NT * DIM) return;
    float v = (i < (long)NU * DIM) ? uf[i] : itf[i - (long)NU * DIM];
    emb[i] = v;
    total[i] = v;
}

__global__ __launch_bounds__(256) void k_ui_deg(const int* __restrict__ src,
                                                const int* __restrict__ dst,
                                                float* deg) {
    long e = (long)blockIdx.x * blockDim.x + threadIdx.x;
    if (e >= NE) return;
    atomicAdd(&deg[src[e]], 1.0f);
    atomicAdd(&deg[NU + dst[e]], 1.0f);
}

__global__ __launch_bounds__(256) void k_dinv(const float* deg, float* dinv) {
    long i = (long)blockIdx.x * blockDim.x + threadIdx.x;
    if (i < NT) dinv[i] = rsqrtf(fmaxf(deg[i], 1.0f));
}

// 4 dims per thread: b128 loads feeding 4 f32 atomics each direction
__global__ __launch_bounds__(256) void k_spmm(const int* __restrict__ src,
                                              const int* __restrict__ dst,
                                              const float* __restrict__ dinv,
                                              const float* __restrict__ emb,
                                              float* gnn) {
    long t = (long)blockIdx.x * blockDim.x + threadIdx.x;
    if (t >= NE * 16) return;
    long e = t >> 4;
    int d0 = (int)(t & 15) << 2;
    int s = src[e];
    int it = NU + dst[e];
    float w = dinv[s] * dinv[it];
    v4f_ xi = *(const v4f_*)&emb[(size_t)it * DIM + d0];
    v4f_ xs = *(const v4f_*)&emb[(size_t)s * DIM + d0];
    float* gs = &gnn[(size_t)s * DIM + d0];
    float* gi = &gnn[(size_t)it * DIM + d0];
#pragma unroll
    for (int q = 0; q < 4; ++q) {
        atomicAdd(gs + q, w * xi[q]);
        atomicAdd(gi + q, w * xs[q]);
    }
}

__global__ __launch_bounds__(256) void k_gc_deg(const int* __restrict__ src,
                                                const int* __restrict__ dst,
                                                float* outdeg, float* indeg) {
    long e = (long)blockIdx.x * blockDim.x + threadIdx.x;
    if (e >= NMPE) return;
    atomicAdd(&outdeg[src[e]], 1.0f);
    atomicAdd(&indeg[dst[e]], 1.0f);
}

__global__ __launch_bounds__(256) void k_gc_scatter(const int* __restrict__ src,
                                                    const int* __restrict__ dst,
                                                    const float* __restrict__ feat,
                                                    const float* __restrict__ outdeg,
                                                    float* z) {
    long t = (long)blockIdx.x * blockDim.x + threadIdx.x;
    if (t >= NMPE * 16) return;
    long e = t >> 4;
    int d0 = (int)(t & 15) << 2;
    int s = src[e];
    float w = rsqrtf(fmaxf(outdeg[s], 1.0f));
    v4f_ x = *(const v4f_*)&feat[(size_t)s * DIM + d0];
    float* zd = &z[(size_t)dst[e] * DIM + d0];
#pragma unroll
    for (int q = 0; q < 4; ++q) atomicAdd(zd + q, w * x[q]);
}

__global__ __launch_bounds__(256) void k_gc_scale(float* z, const float* __restrict__ indeg,
                                                  long n) {
    long i = (long)blockIdx.x * blockDim.x + threadIdx.x;
    if (i >= n * DIM) return;
    z[i] *= rsqrtf(fmaxf(indeg[i >> 6], 1.0f));
}

// ---------------- intent WMMA kernel ----------------
// Per wave: 16-node tile. logits = X(16x64)@intent(64x128) [bf16 WMMA],
// row softmax, out = P(16x128)@intent^T(128x64) [bf16 WMMA],
// emb = gnn + out + emb ; total += emb.
#define IWV 4   // waves per block
__global__ __launch_bounds__(128) void k_intent(float* __restrict__ emb,
                                                const float* __restrict__ gnn,
                                                float* __restrict__ total,
                                                const float* __restrict__ intent,
                                                int count) {
    __shared__ __attribute__((aligned(16))) unsigned short sInt[DIM][NINT];    // (d,j)  16KB
    __shared__ __attribute__((aligned(16))) unsigned short sIntT[NINT][DIM];   // (j,d)  16KB
    __shared__ __attribute__((aligned(16))) unsigned short sX[IWV][16][DIM];   //         8KB
    __shared__ __attribute__((aligned(16))) unsigned short sP[IWV][16][NINT];  //        16KB

    const int tid = threadIdx.x;
    // stage intent (row-major, packed pairs of j)
    for (int i = tid; i < DIM * NINT / 2; i += blockDim.x)
        ((unsigned*)sInt)[i] = packbf2(intent[2 * i], intent[2 * i + 1]);
    // stage intent transposed (j,d), packed pairs of d
    for (int i = tid; i < DIM * NINT / 2; i += blockDim.x) {
        int j = i >> 5;
        int d2 = (i & 31) << 1;
        ((unsigned*)sIntT)[i] = packbf2(intent[(size_t)d2 * NINT + j],
                                        intent[(size_t)(d2 + 1) * NINT + j]);
    }

    const int wave = tid >> 5;
    const int lane = tid & 31;
    const int ntiles = count >> 4;
    const int tile = blockIdx.x * IWV + wave;
    const bool active = tile < ntiles;
    const int node0 = active ? (tile << 4) : 0;

    // stage X tile (16x64 contiguous floats), packed bf16 pairs
    {
        const float* xb = emb + (size_t)node0 * DIM;
        for (int i = lane; i < 16 * DIM / 2; i += 32)
            ((unsigned*)sX[wave])[i] = packbf2(xb[2 * i], xb[2 * i + 1]);
    }
    __syncthreads();

    const int lm = lane & 15;            // A row / B col within tile
    const int lh = (lane >> 4) * 8;      // K sub-offset (and C/D row offset)

    // A fragments for X (K = 0..31 and 32..63): 16B LDS loads
    const unsigned short* xr = &sX[wave][lm][0];
    v16bf a0 = frag16(xr + lh,      xr + lh + 16);
    v16bf a1 = frag16(xr + lh + 32, xr + lh + 48);

    // forward matmul: 8 N-tiles of logits
    v8f acc[8];
    for (int nt = 0; nt < 8; ++nt) {
        const int jc = (nt << 4) + lm;
        const unsigned short* br = &sIntT[jc][0];
        v16bf b0 = frag16(br + lh,      br + lh + 16);
        v16bf b1 = frag16(br + lh + 32, br + lh + 48);
        v8f c = {};
        c = __builtin_amdgcn_wmma_f32_16x16x32_bf16(false, a0, false, b0, (short)0, c, false, false);
        c = __builtin_amdgcn_wmma_f32_16x16x32_bf16(false, a1, false, b1, (short)0, c, false, false);
        acc[nt] = c;
    }

    // softmax over 128 cols per row; row = r + lh, cols spread over 16-lane group x 8 tiles
#pragma unroll
    for (int r = 0; r < 8; ++r) {
        float mx = -1e30f;
#pragma unroll
        for (int nt = 0; nt < 8; ++nt) mx = fmaxf(mx, acc[nt][r]);
        mx = max16(mx);
        float ex[8], s = 0.0f;
#pragma unroll
        for (int nt = 0; nt < 8; ++nt) { ex[nt] = __expf(acc[nt][r] - mx); s += ex[nt]; }
        s = sum16(s);
        const float inv = 1.0f / s;
        const int row = r + lh;
#pragma unroll
        for (int nt = 0; nt < 8; ++nt)
            sP[wave][row][(nt << 4) + lm] = bf16bits(ex[nt] * inv);
    }
    __syncthreads();

    // backward matmul: out(16x64) = P(16x128) @ intent^T(128x64)
    v8f o[4];
#pragma unroll
    for (int i = 0; i < 4; ++i) { v8f zz = {}; o[i] = zz; }
    const unsigned short* pr = &sP[wave][lm][0];
#pragma unroll
    for (int kf = 0; kf < 4; ++kf) {
        v16bf a = frag16(pr + (kf << 5) + lh, pr + (kf << 5) + lh + 16);
#pragma unroll
        for (int nt2 = 0; nt2 < 4; ++nt2) {
            const unsigned short* br = &sInt[(nt2 << 4) + lm][0];   // B(k=j, n=d)=intent[d][j]
            v16bf b = frag16(br + (kf << 5) + lh, br + (kf << 5) + lh + 16);
            o[nt2] = __builtin_amdgcn_wmma_f32_16x16x32_bf16(false, a, false, b, (short)0,
                                                             o[nt2], false, false);
        }
    }

    if (active) {
#pragma unroll
        for (int nt2 = 0; nt2 < 4; ++nt2) {
#pragma unroll
            for (int r = 0; r < 8; ++r) {
                const int m = r + lh;
                const int d = (nt2 << 4) + lm;
                const size_t idx = (size_t)(node0 + m) * DIM + d;
                float nv = gnn[idx] + o[nt2][r] + emb[idx];
                emb[idx] = nv;
                total[idx] += nv;
            }
        }
    }
}

// ---------------- semantic attention (WMMA) ----------------
// w[mp] += sum over 16-node tile of tanh(Z(16x64)@W1(64x128) + b1) @ w2
__global__ __launch_bounds__(128) void k_att_w(const float* __restrict__ z,   // [2][n][64]
                                               const float* __restrict__ W1,  // [64][128]
                                               const float* __restrict__ b1,
                                               const float* __restrict__ w2,
                                               int n, float* wout) {
    __shared__ __attribute__((aligned(16))) unsigned short sW1T[HIDN][DIM];   // 16KB
    __shared__ __attribute__((aligned(16))) unsigned short sZ[IWV][16][DIM];  //  8KB
    const int tid = threadIdx.x;
    for (int i = tid; i < DIM * HIDN / 2; i += blockDim.x) {
        int j = i >> 5;
        int d2 = (i & 31) << 1;
        ((unsigned*)sW1T)[i] = packbf2(W1[(size_t)d2 * HIDN + j],
                                       W1[(size_t)(d2 + 1) * HIDN + j]);
    }
    const int wave = tid >> 5, lane = tid & 31;
    const int ntiles = n >> 4;
    const int tile = blockIdx.x * IWV + wave;
    const bool active = tile < ntiles;
    const int node0 = active ? (tile << 4) : 0;
    const int lm = lane & 15, lh = (lane >> 4) * 8;

    float accmp[2];
#pragma unroll
    for (int mp = 0; mp < 2; ++mp) {
        __syncthreads();                       // sW1T ready (mp==0) / sZ reuse (mp==1)
        const float* zb = z + ((size_t)mp * n + node0) * DIM;
        for (int i = lane; i < 16 * DIM / 2; i += 32)
            ((unsigned*)sZ[wave])[i] = packbf2(zb[2 * i], zb[2 * i + 1]);
        __syncthreads();
        const unsigned short* zr = &sZ[wave][lm][0];
        v16bf a0 = frag16(zr + lh,      zr + lh + 16);
        v16bf a1 = frag16(zr + lh + 32, zr + lh + 48);
        float acc = 0.0f;
        for (int nt = 0; nt < 8; ++nt) {
            const int jc = (nt << 4) + lm;
            const unsigned short* br = &sW1T[jc][0];
            v16bf bb0 = frag16(br + lh,      br + lh + 16);
            v16bf bb1 = frag16(br + lh + 32, br + lh + 48);
            v8f c = {};
            c = __builtin_amdgcn_wmma_f32_16x16x32_bf16(false, a0, false, bb0, (short)0, c, false, false);
            c = __builtin_amdgcn_wmma_f32_16x16x32_bf16(false, a1, false, bb1, (short)0, c, false, false);
            const float bj = b1[jc], wj = w2[jc];
#pragma unroll
            for (int r = 0; r < 8; ++r) acc += tanhf(c[r] + bj) * wj;
        }
        accmp[mp] = acc;
    }
#pragma unroll
    for (int mp = 0; mp < 2; ++mp) {
        float a = sum32(accmp[mp]);
        if (lane == 0 && active) atomicAdd(&wout[mp], a);
    }
}

__global__ void k_beta(const float* watt, float* beta) {
    int s = threadIdx.x;      // 0 = user side, 1 = item side
    if (s >= 2) return;
    const float n = 100000.0f;
    float w0 = watt[s * 2 + 0] / n, w1 = watt[s * 2 + 1] / n;
    float m = fmaxf(w0, w1);
    float e0 = __expf(w0 - m), e1 = __expf(w1 - m);
    float inv = 1.0f / (e0 + e1);
    beta[s * 2 + 0] = e0 * inv;
    beta[s * 2 + 1] = e1 * inv;
}

__global__ __launch_bounds__(256) void k_h2(const float* __restrict__ z,
                                            const float* __restrict__ beta,
                                            float* h2, long n) {
    long i = (long)blockIdx.x * blockDim.x + threadIdx.x;
    if (i >= n * DIM) return;
    h2[i] = beta[0] * z[i] + beta[1] * z[n * DIM + i];
}

// ---------------- SSL ----------------
__global__ __launch_bounds__(32) void k_ssl_prep(const float* __restrict__ h2,
                                                 const float* __restrict__ aSl,
                                                 const int* __restrict__ idx,
                                                 float* e1n, float* e2n,
                                                 unsigned short* e1b, unsigned short* e2b) {
    const int b = blockIdx.x, lane = threadIdx.x;
    const int node = idx[b];
    float x0 = h2[(size_t)node * DIM + lane], x1 = h2[(size_t)node * DIM + lane + 32];
    float y0 = aSl[(size_t)node * DIM + lane], y1 = aSl[(size_t)node * DIM + lane + 32];
    float sx = sum32(x0 * x0 + x1 * x1);
    float sy = sum32(y0 * y0 + y1 * y1);
    const float ix = rsqrtf(sx), iy = rsqrtf(sy);
    x0 *= ix; x1 *= ix; y0 *= iy; y1 *= iy;
    const size_t o = (size_t)b * DIM + lane;
    e1n[o] = x0; e1n[o + 32] = x1;
    e2n[o] = y0; e2n[o + 32] = y1;
    e1b[o] = bf16bits(x0); e1b[o + 32] = bf16bits(x1);
    e2b[o] = bf16bits(y0); e2b[o + 32] = bf16bits(y1);
}

// one 16x16 similarity tile per wave; rowsum[m] += sum_n exp(2*dot)
__global__ __launch_bounds__(128) void k_ssl_mm(const unsigned short* __restrict__ e1b,
                                                const unsigned short* __restrict__ e2b,
                                                float* rowsum) {
    const int wave = threadIdx.x >> 5, lane = threadIdx.x & 31;
    const int tileId = blockIdx.x * 4 + wave;          // 65536 tiles
    const int mt = tileId >> 8, nt = tileId & 255;
    const int lm = lane & 15, lh = (lane >> 4) * 8;
    const int row0 = mt << 4, col0 = nt << 4;
    const unsigned short* r1 = e1b + (size_t)(row0 + lm) * DIM;
    const unsigned short* r2 = e2b + (size_t)(col0 + lm) * DIM;
    v16bf a0 = frag16(r1 + lh,      r1 + lh + 16);
    v16bf a1 = frag16(r1 + lh + 32, r1 + lh + 48);
    v16bf b0 = frag16(r2 + lh,      r2 + lh + 16);
    v16bf b1 = frag16(r2 + lh + 32, r2 + lh + 48);
    v8f c = {};
    c = __builtin_amdgcn_wmma_f32_16x16x32_bf16(false, a0, false, b0, (short)0, c, false, false);
    c = __builtin_amdgcn_wmma_f32_16x16x32_bf16(false, a1, false, b1, (short)0, c, false, false);
#pragma unroll
    for (int r = 0; r < 8; ++r) {
        float v = sum16(__expf(2.0f * c[r]));  // unit-vector dot in [-1,1]: exp safe
        if (lm == 0) atomicAdd(&rowsum[row0 + r + lh], v);
    }
}

__global__ __launch_bounds__(32) void k_ssl_loss(const float* __restrict__ e1n,
                                                 const float* __restrict__ e2n,
                                                 const float* __restrict__ rowsum,
                                                 float* loss) {
    const int b = blockIdx.x, lane = threadIdx.x;
    const size_t o = (size_t)b * DIM + lane;
    float p = sum32(e1n[o] * e2n[o] + e1n[o + 32] * e2n[o + 32]);
    if (lane == 0) {
        float pos = 2.0f * p;                 // /0.5 temperature
        float lse = logf(rowsum[b]);
        atomicAdd(loss, -(pos - lse) / (float)BSZ);
    }
}

// ---------------- output gather ----------------
__global__ __launch_bounds__(256) void k_out(const float* __restrict__ total,
                                             const float* __restrict__ h2u,
                                             const float* __restrict__ h2i,
                                             const int* __restrict__ uidx,
                                             const int* __restrict__ iidx,
                                             const int* __restrict__ nidx,
                                             float* out) {
    long t = (long)blockIdx.x * blockDim.x + threadIdx.x;
    if (t >= (long)BSZ * DIM) return;
    const int b = (int)(t >> 6), d = (int)(t & 63);
    const int u = uidx[b], it = iidx[b], ng = nidx[b];
    out[t] = 0.5f * (total[(size_t)u * DIM + d] + h2u[(size_t)u * DIM + d]);
    out[(long)BSZ * DIM + t] =
        0.5f * (total[(size_t)(NU + it) * DIM + d] + h2i[(size_t)it * DIM + d]);
    out[2L * BSZ * DIM + t] =
        0.5f * (total[(size_t)(NU + ng) * DIM + d] + h2i[(size_t)ng * DIM + d]);
}

// ---------------- host orchestration ----------------
static inline int gb(long n, int bs) { return (int)((n + bs - 1) / bs); }

extern "C" void kernel_launch(void* const* d_in, const int* in_sizes, int n_in,
                              void* d_out, int out_size, void* d_ws, size_t ws_size,
                              hipStream_t stream) {
    const float* user_feat   = (const float*)d_in[0];
    const float* item_feat   = (const float*)d_in[1];
    const float* user_intent = (const float*)d_in[2];
    const float* item_intent = (const float*)d_in[3];
    const float* sa_u_W1     = (const float*)d_in[4];
    const float* sa_u_b1     = (const float*)d_in[5];
    const float* sa_u_w2     = (const float*)d_in[6];
    const float* sa_i_W1     = (const float*)d_in[7];
    const float* sa_i_b1     = (const float*)d_in[8];
    const float* sa_i_w2     = (const float*)d_in[9];
    const int*   ui_src      = (const int*)d_in[10];
    const int*   ui_dst      = (const int*)d_in[11];
    const int*   mp_user     = (const int*)d_in[12];   // [2][2][NMPE]
    const int*   mp_item     = (const int*)d_in[13];
    const int*   user_idx    = (const int*)d_in[14];
    const int*   item_idx    = (const int*)d_in[15];
    const int*   neg_idx     = (const int*)d_in[16];
    float* out = (float*)d_out;

    float* W = (float*)d_ws;
    float* deg    = W + OFF_DEG;
    float* dinv   = W + OFF_DINV;
    float* emb    = W + OFF_EMB;
    float* total  = W + OFF_TOTAL;
    float* gnn    = W + OFF_GNN;
    float* h2u    = W + OFF_GNN;                       // reuse after layers
    float* h2i    = W + OFF_GNN + (long)NU * DIM;
    float* zu     = W + OFF_Z;                         // [2][NU][64]
    float* zi     = W + OFF_Z + 2L * NU * DIM;         // [2][NI][64]
    float* e1n    = W + OFF_E1N;
    float* e2n    = W + OFF_E2N;
    unsigned short* e1b = (unsigned short*)(W + OFF_E1B);
    unsigned short* e2b = (unsigned short*)(W + OFF_E2B);
    float* rowsum = W + OFF_RS;
    float* watt   = W + OFF_WATT;
    float* beta   = W + OFF_BETA;

    // --- LightGCN-with-intent layers ---
    k_zero<<<gb(NT, 256), 256, 0, stream>>>(deg, NT);
    k_init_emb<<<gb((long)NT * DIM, 256), 256, 0, stream>>>(user_feat, item_feat, emb, total);
    k_ui_deg<<<gb(NE, 256), 256, 0, stream>>>(ui_src, ui_dst, deg);
    k_dinv<<<gb(NT, 256), 256, 0, stream>>>(deg, dinv);

    const int intentBlocksU = gb((NU / 16), IWV);
    const int intentBlocksI = gb((NI / 16), IWV);
    for (int layer = 0; layer < NLAYERS; ++layer) {
        k_zero<<<gb((long)NT * DIM, 256), 256, 0, stream>>>(gnn, (long)NT * DIM);
        k_spmm<<<gb(NE * 16, 256), 256, 0, stream>>>(ui_src, ui_dst, dinv, emb, gnn);
        k_intent<<<intentBlocksU, 128, 0, stream>>>(emb, gnn, total, user_intent, NU);
        k_intent<<<intentBlocksI, 128, 0, stream>>>(emb + (long)NU * DIM, gnn + (long)NU * DIM,
                                                    total + (long)NU * DIM, item_intent, NI);
    }

    // --- metapath graph convolutions ---
    k_zero<<<gb(4L * NU * DIM, 256), 256, 0, stream>>>(zu, 4L * NU * DIM);  // zu + zi contiguous
    for (int side = 0; side < 2; ++side) {
        const int* mp = (side == 0) ? mp_user : mp_item;
        const float* feat = (side == 0) ? user_feat : item_feat;
        float* z = (side == 0) ? zu : zi;
        const int n = (side == 0) ? NU : NI;
        for (int m = 0; m < 2; ++m) {
            const int* src = mp + ((long)m * 2 + 0) * NMPE;
            const int* dst = mp + ((long)m * 2 + 1) * NMPE;
            float* outdeg = deg;            // reuse
            float* indeg  = deg + n;
            float* zs = z + (long)m * n * DIM;
            k_zero<<<gb(2L * n, 256), 256, 0, stream>>>(outdeg, 2L * n);
            k_gc_deg<<<gb(NMPE, 256), 256, 0, stream>>>(src, dst, outdeg, indeg);
            k_gc_scatter<<<gb(NMPE * 16, 256), 256, 0, stream>>>(src, dst, feat, outdeg, zs);
            k_gc_scale<<<gb((long)n * DIM, 256), 256, 0, stream>>>(zs, indeg, n);
        }
    }

    // --- semantic attention ---
    k_zero<<<1, 8, 0, stream>>>(watt, 8);
    k_att_w<<<gb(NU / 16, IWV), 128, 0, stream>>>(zu, sa_u_W1, sa_u_b1, sa_u_w2, NU, watt);
    k_att_w<<<gb(NI / 16, IWV), 128, 0, stream>>>(zi, sa_i_W1, sa_i_b1, sa_i_w2, NI, watt + 2);
    k_beta<<<1, 2, 0, stream>>>(watt, beta);
    k_h2<<<gb((long)NU * DIM, 256), 256, 0, stream>>>(zu, beta, h2u, NU);
    k_h2<<<gb((long)NI * DIM, 256), 256, 0, stream>>>(zi, beta + 2, h2i, NI);

    // --- SSL losses ---
    float* lossPtr = out + 3L * BSZ * DIM;
    k_zero<<<1, 1, 0, stream>>>(lossPtr, 1);

    // users
    k_zero<<<gb(BSZ, 256), 256, 0, stream>>>(rowsum, BSZ);
    k_ssl_prep<<<BSZ, 32, 0, stream>>>(h2u, total, user_idx, e1n, e2n, e1b, e2b);
    k_ssl_mm<<<(BSZ / 16) * (BSZ / 16) / 4, 128, 0, stream>>>(e1b, e2b, rowsum);
    k_ssl_loss<<<BSZ, 32, 0, stream>>>(e1n, e2n, rowsum, lossPtr);
    // items
    k_zero<<<gb(BSZ, 256), 256, 0, stream>>>(rowsum, BSZ);
    k_ssl_prep<<<BSZ, 32, 0, stream>>>(h2i, total + (long)NU * DIM, item_idx, e1n, e2n, e1b, e2b);
    k_ssl_mm<<<(BSZ / 16) * (BSZ / 16) / 4, 128, 0, stream>>>(e1b, e2b, rowsum);
    k_ssl_loss<<<BSZ, 32, 0, stream>>>(e1n, e2n, rowsum, lossPtr);

    // --- final gathers ---
    k_out<<<gb((long)BSZ * DIM, 256), 256, 0, stream>>>(total, h2u, h2i,
                                                        user_idx, item_idx, neg_idx, out);
}
